// NerfModel_42537356100386
// MI455X (gfx1250) — matrix operand order repaired and checked by reading, compiled, not run on hardware
//
#include <hip/hip_runtime.h>
#include <stdint.h>

// ---------------- constants ----------------
#define NSAMP 262144
#define CHUNKS (NSAMP / 64)          // 64 rows per workgroup chunk (4 waves x 16 rows)

// element offsets (bf16) of each layer's swizzled weight tiles in d_ws / LDS
constexpr int ELOFF_[11] = {0, 8192, 24576, 40960, 57344, 81920,
                            98304, 114688, 133120, 143360, 144384};
constexpr int KT_[10]   = {2, 4, 4, 4, 6, 4, 4, 4, 5, 2};
constexpr int FI_[10]   = {60, 128, 128, 128, 188, 128, 128, 128, 152, 64};
constexpr int FO_[10]   = {128, 128, 128, 128, 128, 128, 128, 129, 64, 3};
constexpr int BOFF_[11] = {0, 128, 256, 384, 512, 640, 768, 896, 1040, 1104, 1120};

#define W_BYTES   (144384 * 2)            // 288768 B of bf16 weights
#define B_BYTES   (1120 * 4)              // 4480 B of f32 biases
#define WB_BYTES  (W_BYTES + B_BYTES)     // 293248 B staged global->LDS
#define ACT_OFF   WB_BYTES                // per-wave activation region after weights
#define ACT_PER_WAVE 7168                 // embx 2048 + embd 1024 + hbuf 4096
#define SMEM_BYTES (WB_BYTES + 4 * ACT_PER_WAVE)   // 321920 < 327680 (320KB)

typedef __attribute__((ext_vector_type(16))) __bf16 bvec16;
typedef __attribute__((ext_vector_type(8)))  float  fvec8;
typedef int vi4 __attribute__((vector_size(16)));
typedef __attribute__((address_space(1))) vi4* gvi4p;   // global v4i*
typedef __attribute__((address_space(3))) vi4* lvi4p;   // LDS v4i*

struct PW { const float* W[10]; const float* b[10]; };

// ---------------- weight pre-swizzle (fp32 -> bf16, WMMA B layout) ----------------
__global__ void nerf_prep_weights(PW p, char* __restrict__ ws) {
  int idx = blockIdx.x * blockDim.x + threadIdx.x;
  if (idx < 144384) {
    int l = 0;
    while (idx >= ELOFF_[l + 1]) ++l;
    int r      = idx - ELOFF_[l];
    int tile   = r >> 9;            // 512 elements per 32x16 tile
    int within = r & 511;
    int lane   = within >> 4;       // 16 contiguous bf16 per lane
    int m      = within & 15;
    int ktile  = tile % KT_[l];
    int ntile  = tile / KT_[l];
    int nl     = lane & 15;
    int kl     = (m < 8) ? ((lane >> 4) << 3) + m : 16 + ((lane >> 4) << 3) + (m - 8);   // lanes 16-31 hold K {8..15, 24..31}
    int k = ktile * 32 + kl;
    int n = ntile * 16 + nl;
    float v = 0.0f;
    if (k < FI_[l] && n < FO_[l]) v = p.W[l][k * FO_[l] + n];
    ((__bf16*)ws)[idx] = (__bf16)v;
  } else if (idx < 144384 + 1120) {
    int bi = idx - 144384;
    int l = 0;
    while (bi >= BOFF_[l + 1]) ++l;
    int n = bi - BOFF_[l];
    float v = (n < FO_[l]) ? p.b[l][n] : 0.0f;
    ((float*)(ws + W_BYTES))[bi] = v;
  }
}

// ---------------- device helpers ----------------
__device__ __forceinline__ bvec16 ld2x16(const char* p0, const char* p1) {
  union { uint4 u[2]; bvec16 v; } t;
  t.u[0] = *(const uint4*)p0;
  t.u[1] = *(const uint4*)p1;
  return t.v;
}

__device__ __forceinline__ fvec8 zero8() {
  return (fvec8){0.f, 0.f, 0.f, 0.f, 0.f, 0.f, 0.f, 0.f};
}

// D = A*B + D over KT k-tiles and NT n-tiles.  A read row-major bf16 from LDS,
// first SPLIT k-tiles from (a0,s0), rest from (a1,s1).  B tiles pre-swizzled in LDS.
template <int KT, int NT, int SPLIT>
__device__ __forceinline__ void mm_acc(fvec8* acc,
                                       const char* a0, int s0,
                                       const char* a1, int s1,
                                       const char* wb, int lane) {
  const int row  = lane & 15;
  const int half = lane >> 4;
#pragma unroll
  for (int kt = 0; kt < KT; ++kt) {
    const char* ab = (kt < SPLIT) ? a0 : a1;
    const int   st = (kt < SPLIT) ? s0 : s1;
    const int   k0 = (kt < SPLIT) ? kt * 32 : (kt - SPLIT) * 32;
    const char* pa = ab + row * st + k0 * 2 + half * 16;
    bvec16 a = ld2x16(pa, pa + 32);      // lanes 0-15: K {k0..k0+7, k0+16..k0+23}
#pragma unroll
    for (int n = 0; n < NT; ++n) {
      const char* pb = wb + (n * KT + kt) * 1024 + lane * 32;
      bvec16 b = ld2x16(pb, pb + 16);
      acc[n] = __builtin_amdgcn_wmma_f32_16x16x32_bf16(
          false, a, false, b, (short)0, acc[n], false, false);
    }
  }
}

// write a 16x16 f32 tile (D layout) back to a row-major bf16 LDS buffer
__device__ __forceinline__ void store_tile(char* outb, int ostride, int n_tile,
                                           const fvec8& acc, const float* bias,
                                           int lane, bool dorelu) {
  const int nl   = lane & 15;
  const int half = lane >> 4;
  const int col  = n_tile * 16 + nl;
  const float bv = bias[n_tile * 16 + nl];
#pragma unroll
  for (int v = 0; v < 8; ++v) {
    float x = acc[v] + bv;
    if (dorelu) x = fmaxf(x, 0.0f);
    *(__bf16*)(outb + (v + 8 * half) * ostride + col * 2) = (__bf16)x;
  }
}

// ---------------- main fused NeRF kernel ----------------
extern "C" __global__ void __launch_bounds__(128, 1)
nerf_main(const float* __restrict__ o, const float* __restrict__ dvec,
          const char* __restrict__ ws, float* __restrict__ out) {
  extern __shared__ char smem[];
  const int tid  = threadIdx.x;
  const int w    = tid >> 5;
  const int lane = tid & 31;

  // ---- stage all swizzled bf16 weights + f32 biases into LDS via the CDNA5
  //      async global->LDS path (ASYNCcnt), once per workgroup ----
  {
#if __has_builtin(__builtin_amdgcn_global_load_async_to_lds_b128)
    for (int i = tid * 16; i < WB_BYTES; i += 128 * 16) {
      __builtin_amdgcn_global_load_async_to_lds_b128(
          (gvi4p)(uintptr_t)(ws + i),
          (lvi4p)(uintptr_t)(smem + i),
          0, 0);
    }
#else
    const unsigned lbase = (unsigned)(uintptr_t)smem;   // DS-domain byte offset
    for (int i = tid * 16; i < WB_BYTES; i += 128 * 16) {
      unsigned loff = lbase + (unsigned)i;
      unsigned goff = (unsigned)i;
      asm volatile("global_load_async_to_lds_b128 %0, %1, %2"
                   :: "v"(loff), "v"(goff), "s"(ws)
                   : "memory");
    }
#endif
#if __has_builtin(__builtin_amdgcn_s_wait_asynccnt)
    __builtin_amdgcn_s_wait_asynccnt(0);
#else
    asm volatile("s_wait_asynccnt 0x0" ::: "memory");
#endif
  }
  __syncthreads();

  const float* bias = (const float*)(smem + W_BYTES);
  char* act  = smem + ACT_OFF + w * ACT_PER_WAVE;
  char* embx = act;              // 16 x 64 bf16, stride 128B (cols 60-63 pad)
  char* embd = act + 2048;       // 16 x 32 bf16, stride  64B (cols 24-31 pad)
  char* hbuf = act + 3072;       // 16 x 128 bf16, stride 256B
  // byte offsets of each layer's weight tiles inside LDS
  const int WOFF[10] = {0, 16384, 49152, 81920, 114688, 163840,
                        196608, 229376, 266240, 286720};

  const int half = lane >> 4;

  for (int chunk = blockIdx.x; chunk < CHUNKS; chunk += gridDim.x) {
    const int rowbase = chunk * 64 + w * 16;

    // ---- positional encodings (bf16 into LDS, zero padded) ----
#pragma unroll
    for (int i = 0; i < 15; ++i) {           // 480 sin/cos pairs for emb_x
      int s = lane + 32 * i;
      int r = s / 30, rem = s % 30, dim = rem / 10, j = rem % 10;
      float x   = o[(rowbase + r) * 3 + dim];
      float ang = x * (float)(1 << j);
      __bf16* pp = (__bf16*)(embx + r * 128 + (dim * 20 + j * 2) * 2);
      pp[0] = (__bf16)__sinf(ang);
      pp[1] = (__bf16)__cosf(ang);
    }
#pragma unroll
    for (int i = 0; i < 2; ++i) {            // zero emb_x cols 60..63
      int s = lane + 32 * i;
      ((__bf16*)(embx + (s >> 2) * 128))[60 + (s & 3)] = (__bf16)0.0f;
    }
#pragma unroll
    for (int i = 0; i < 6; ++i) {            // 192 sin/cos pairs for emb_d
      int s = lane + 32 * i;
      int r = s / 12, rem = s % 12, dim = rem / 4, j = rem % 4;
      float x   = dvec[(rowbase + r) * 3 + dim];
      float ang = x * (float)(1 << j);
      __bf16* pp = (__bf16*)(embd + r * 64 + (dim * 8 + j * 2) * 2);
      pp[0] = (__bf16)__sinf(ang);
      pp[1] = (__bf16)__cosf(ang);
    }
#pragma unroll
    for (int i = 0; i < 4; ++i) {            // zero emb_d cols 24..31
      int s = lane + 32 * i;
      ((__bf16*)(embd + (s >> 3) * 64))[24 + (s & 7)] = (__bf16)0.0f;
    }

    // ---- L0: emb_x(64) -> 128, relu ----
    {
      fvec8 acc[8];
#pragma unroll
      for (int n = 0; n < 8; ++n) acc[n] = zero8();
      mm_acc<2, 8, 2>(acc, embx, 128, embx, 128, smem + WOFF[0], lane);
#pragma unroll
      for (int n = 0; n < 8; ++n) store_tile(hbuf, 256, n, acc[n], bias + BOFF_[0], lane, true);
    }
    // ---- L1..L3: 128 -> 128, relu ----
#pragma unroll
    for (int l = 1; l <= 3; ++l) {
      fvec8 acc[8];
#pragma unroll
      for (int n = 0; n < 8; ++n) acc[n] = zero8();
      mm_acc<4, 8, 4>(acc, hbuf, 256, hbuf, 256, smem + WOFF[l], lane);
#pragma unroll
      for (int n = 0; n < 8; ++n) store_tile(hbuf, 256, n, acc[n], bias + BOFF_[l], lane, true);
    }
    // ---- L4: concat(h, emb_x) = 192 -> 128, relu ----
    {
      fvec8 acc[8];
#pragma unroll
      for (int n = 0; n < 8; ++n) acc[n] = zero8();
      mm_acc<6, 8, 4>(acc, hbuf, 256, embx, 128, smem + WOFF[4], lane);
#pragma unroll
      for (int n = 0; n < 8; ++n) store_tile(hbuf, 256, n, acc[n], bias + BOFF_[4], lane, true);
    }
    // ---- L5, L6: 128 -> 128, relu ----
#pragma unroll
    for (int l = 5; l <= 6; ++l) {
      fvec8 acc[8];
#pragma unroll
      for (int n = 0; n < 8; ++n) acc[n] = zero8();
      mm_acc<4, 8, 4>(acc, hbuf, 256, hbuf, 256, smem + WOFF[l], lane);
#pragma unroll
      for (int n = 0; n < 8; ++n) store_tile(hbuf, 256, n, acc[n], bias + BOFF_[l], lane, true);
    }
    // ---- L7: 128 -> 129 (h linear, sigma=relu(col 128)) ----
    {
      fvec8 acc[9];
#pragma unroll
      for (int n = 0; n < 9; ++n) acc[n] = zero8();
      mm_acc<4, 9, 4>(acc, hbuf, 256, hbuf, 256, smem + WOFF[7], lane);
#pragma unroll
      for (int n = 0; n < 8; ++n) store_tile(hbuf, 256, n, acc[n], bias + BOFF_[7], lane, false);
      const int nl = lane & 15;
      if (nl == 0) {                           // lanes 0 and 16 hold sigma column
        float bv = bias[BOFF_[7] + 128];
#pragma unroll
        for (int v = 0; v < 8; ++v) {
          float s = fmaxf(acc[8][v] + bv, 0.0f);
          out[3 * NSAMP + rowbase + v + 8 * half] = s;
        }
      }
    }
    // ---- L8: concat(h, emb_d) = 160 -> 64, relu (output into embx region) ----
    {
      fvec8 acc[4];
#pragma unroll
      for (int n = 0; n < 4; ++n) acc[n] = zero8();
      mm_acc<5, 4, 4>(acc, hbuf, 256, embd, 64, smem + WOFF[8], lane);
#pragma unroll
      for (int n = 0; n < 4; ++n) store_tile(embx, 128, n, acc[n], bias + BOFF_[8], lane, true);
    }
    // ---- L9: 64 -> 3, sigmoid -> global rgb ----
    {
      fvec8 acc[1];
      acc[0] = zero8();
      mm_acc<2, 1, 2>(acc, embx, 128, embx, 128, smem + WOFF[9], lane);
      const int nl = lane & 15;
      if (nl < 3) {
        float bv = bias[BOFF_[9] + nl];
#pragma unroll
        for (int v = 0; v < 8; ++v) {
          float x  = acc[0][v] + bv;
          float sg = 1.0f / (1.0f + __expf(-x));
          out[(rowbase + v + 8 * half) * 3 + nl] = sg;
        }
      }
    }
  }
}

// ---------------- host launcher ----------------
extern "C" void kernel_launch(void* const* d_in, const int* in_sizes, int n_in,
                              void* d_out, int out_size, void* d_ws, size_t ws_size,
                              hipStream_t stream) {
  const float* o  = (const float*)d_in[0];
  const float* dv = (const float*)d_in[1];
  PW p;
  for (int i = 0; i < 10; ++i) {
    p.W[i] = (const float*)d_in[2 + 2 * i];
    p.b[i] = (const float*)d_in[3 + 2 * i];
  }
  char*  ws  = (char*)d_ws;
  float* out = (float*)d_out;

  (void)hipFuncSetAttribute((const void*)nerf_main,
                            hipFuncAttributeMaxDynamicSharedMemorySize, SMEM_BYTES);

  nerf_prep_weights<<<(144384 + 1120 + 255) / 256, 256, 0, stream>>>(p, ws);
  nerf_main<<<512, 128, SMEM_BYTES, stream>>>(o, dv, ws, out);
}
